// IndustrialSSMCore_76845554860140
// MI455X (gfx1250) — matrix operand order repaired and checked by reading, not compiled, not run on hardware
//
#include <hip/hip_runtime.h>
#include <hip/hip_bf16.h>
#include <math.h>

// ---------------- model dims ----------------
#define DIM      1024
#define D_INNER  2048
#define D_STATE  16
#define D_CONV   4
#define DT_RANK  64
#define BATCH    2
#define SEQ      1024
#define NTOK     (BATCH * SEQ)           // 2048
#define XPROJ_N  (DT_RANK + 2 * D_STATE) // 96

typedef __attribute__((ext_vector_type(16))) _Float16     v16h;
typedef __attribute__((ext_vector_type(8)))  _Float16     h8;
typedef __attribute__((ext_vector_type(8)))  float        v8f;
typedef __attribute__((ext_vector_type(4)))  unsigned int u32x4;
typedef __attribute__((ext_vector_type(8)))  int          i32x8;
typedef __attribute__((ext_vector_type(4)))  int          i32x4;

// Tensor Data Mover availability (this toolchain: 6-arg builtin)
#if defined(__has_builtin)
#  if __has_builtin(__builtin_amdgcn_tensor_load_to_lds) && __has_builtin(__builtin_amdgcn_s_wait_tensorcnt)
#    define USE_TDM 1
#  endif
#endif
#ifndef USE_TDM
#  define USE_TDM 0
#endif

// ---------------------------------------------------------------------------
// f32 -> f16 bulk conversion (8 elements/thread, n must be a multiple of 8)
// ---------------------------------------------------------------------------
__global__ __launch_bounds__(256)
void cvt_f32_f16(const float* __restrict__ src, _Float16* __restrict__ dst, size_t n)
{
    const size_t i = ((size_t)blockIdx.x * blockDim.x + threadIdx.x) * 8;
    if (i >= n) return;
    const float4 a = *(const float4*)(src + i);
    const float4 b = *(const float4*)(src + i + 4);
    h8 o;
    o[0] = (_Float16)a.x; o[1] = (_Float16)a.y; o[2] = (_Float16)a.z; o[3] = (_Float16)a.w;
    o[4] = (_Float16)b.x; o[5] = (_Float16)b.y; o[6] = (_Float16)b.z; o[7] = (_Float16)b.w;
    *(h8*)(dst + i) = o;
}

// ---------------------------------------------------------------------------
// WMMA fragment helpers (contiguous K order, shared by A and B)
// A fragment (16x32, MxK): lane m = lane%16, lo = (lane>=16)?8:0
//   halves i <-> K = k0+2*lo+i (16 contiguous)
// B fragment (32x16, KxN): lane n = lane%16, kb = (lane>=16)?16:0
//   halves i <-> K = k0+kb+i (16 contiguous)
// ---------------------------------------------------------------------------
__device__ __forceinline__ v16h frag_a_f16(const _Float16* __restrict__ row, int k0, int lo) {
    const h8 c0 = *(const h8*)(row + k0 + 2 * lo);
    const h8 c1 = *(const h8*)(row + k0 + 2 * lo + 8);
    v16h r;
    #pragma unroll
    for (int i = 0; i < 8; ++i) { r[i] = c0[i]; r[8 + i] = c1[i]; }
    return r;
}

__device__ __forceinline__ v16h frag_b_16(const _Float16* p) {
    const h8 c0 = *(const h8*)(p);
    const h8 c1 = *(const h8*)(p + 8);
    v16h r;
    #pragma unroll
    for (int i = 0; i < 8; ++i) { r[i] = c0[i]; r[8 + i] = c1[i]; }
    return r;
}

__device__ __forceinline__ float softplus_eps(float x) {
    return (x > 20.0f ? x : log1pf(__expf(x))) + 1e-4f;
}

// ---------------------------------------------------------------------------
// WMMA GEMM:  C[M,N] = A[M,K] @ W[N,K]^T   (f16 operands, f32 accumulate)
// Block = 128 threads = 4 waves stacked along M; wave strip = 16(M) x 16*NT(N).
// All 4 waves share the weight panel -> double-buffered LDS staging:
//   USE_TDM: wave 0 issues TENSOR_LOAD_TO_LDS for panel k+32 while all waves
//            run WMMAs on panel k; s_wait_tensorcnt 0 + barrier to flip.
//   else:    next-panel loads hoisted to regs before the WMMAs, ds_store after.
// Grid: x = M/64, y = N/(16*NT) (exact fit by launch choice).
// EPI: 0 = plain store, 1 = softplus(acc + bias[n]) + 1e-4.
// ---------------------------------------------------------------------------
template<int NT, int EPI>
__global__ __launch_bounds__(128)
void gemm_wmma(const _Float16* __restrict__ A, const _Float16* __restrict__ W,
               float* __restrict__ C, const float* __restrict__ bias,
               int K, int lda, int ldw, int ldc)
{
    constexpr int NB = 16 * NT;                 // block N width
    constexpr int CH = NB / 32;                 // 16B staging chunks per thread
    __shared__ _Float16 sB[2][NB * 32];         // [buf][row (N) * 32 (K halves)]

    const int lane = threadIdx.x & 31;
    const int wave = threadIdx.x >> 5;
    const int m0   = blockIdx.x * 64 + wave * 16;
    const int n0   = blockIdx.y * NB;

    const int mrow = lane & 15;
    const int lo   = (lane >> 4) * 8;
    const int kb   = (lane >> 4) * 16;

    const _Float16* __restrict__ Arow = A + (size_t)(m0 + mrow) * lda;

    v8f acc[NT];
    #pragma unroll
    for (int j = 0; j < NT; ++j) acc[j] = v8f{};

#if USE_TDM
    // Tensor DMA descriptor per CDNA5 ISA:
    // 2D tile: tile_dim0 = 32 halves (contiguous K), tile_dim1 = NB rows,
    // tensor_dim0_stride = ldw halves, data_size = 2B, LDS dest = sB[buf].
    auto tdm_issue = [&](int buf, int k0) {
        const unsigned long long g =
            (unsigned long long)(uintptr_t)(W + (size_t)n0 * ldw + k0);
        u32x4 g0;
        g0[0] = 1u;                                                // count=1
        g0[1] = (unsigned int)(buf * (NB * 64));                   // lds byte addr (sB @ 0)
        g0[2] = (unsigned int)g;                                   // global addr lo
        g0[3] = (unsigned int)((g >> 32) & 0x1FFFFFFu) | 0x80000000u; // addr hi | type=2
        i32x8 g1;
        g1[0] = (int)(1u << 16);                 // workgroup_mask=0, data_size=1 (2B)
        g1[1] = 0;                               // tensor_dim0 = 1<<20 -> low16 = 0
        g1[2] = 0x10;                            // dim0 hi16 = 0x10, dim1 low16 = 0
        g1[3] = (32 << 16) | 0x10;               // dim1 hi16 = 0x10, tile_dim0 = 32
        g1[4] = NB;                              // tile_dim1 = NB, tile_dim2 = 0
        g1[5] = ldw;                             // tensor_dim0_stride lo32 (halves)
        g1[6] = 0;                               // stride hi, dim1_stride lo = 0
        g1[7] = 0;
        const i32x4 z4 = {};
        const i32x8 z8 = {};
        __builtin_amdgcn_tensor_load_to_lds(g0, g1, z4, z4, z8, 0);
    };

    if (wave == 0) {
        tdm_issue(0, 0);
        __builtin_amdgcn_s_wait_tensorcnt((short)0);
    }
    __syncthreads();

    for (int k0 = 0; k0 < K; k0 += 32) {
        const int cur = (k0 >> 5) & 1;
        if (k0 + 32 < K) {
            if (wave == 0) tdm_issue(cur ^ 1, k0 + 32);   // DMA next panel
            __builtin_prefetch(Arow + k0 + 32, 0, 1);
        }
        const v16h a = frag_a_f16(Arow, k0, lo);
        #pragma unroll
        for (int j = 0; j < NT; ++j) {
            const v16h b = frag_b_16(&sB[cur][(16 * j + mrow) * 32 + kb]);
            acc[j] = __builtin_amdgcn_wmma_f32_16x16x32_f16(
                false, a, false, b, (short)0, acc[j], false, false);
        }
        if (wave == 0) __builtin_amdgcn_s_wait_tensorcnt((short)0);
        __syncthreads();
    }
#else
    // fallback: software-pipelined register staging
    #pragma unroll
    for (int c = 0; c < CH; ++c) {
        const int chunk = threadIdx.x + 128 * c;   // 16B chunks, 4 per row
        const int row   = chunk >> 2;
        const int koff  = (chunk & 3) * 8;
        *(h8*)(&sB[0][row * 32 + koff]) =
            *(const h8*)(W + (size_t)(n0 + row) * ldw + koff);
    }
    __syncthreads();

    for (int k0 = 0; k0 < K; k0 += 32) {
        const int cur  = (k0 >> 5) & 1;
        const bool more = (k0 + 32 < K);
        h8 pre[CH];
        if (more) {                                // issue loads BEFORE the WMMAs
            #pragma unroll
            for (int c = 0; c < CH; ++c) {
                const int chunk = threadIdx.x + 128 * c;
                const int row   = chunk >> 2;
                const int koff  = (chunk & 3) * 8;
                pre[c] = *(const h8*)(W + (size_t)(n0 + row) * ldw + k0 + 32 + koff);
            }
            __builtin_prefetch(Arow + k0 + 32, 0, 1);
        }
        const v16h a = frag_a_f16(Arow, k0, lo);
        #pragma unroll
        for (int j = 0; j < NT; ++j) {
            const v16h b = frag_b_16(&sB[cur][(16 * j + mrow) * 32 + kb]);
            acc[j] = __builtin_amdgcn_wmma_f32_16x16x32_f16(
                false, a, false, b, (short)0, acc[j], false, false);
        }
        if (more) {                                // land stores AFTER the WMMAs
            #pragma unroll
            for (int c = 0; c < CH; ++c) {
                const int chunk = threadIdx.x + 128 * c;
                const int row   = chunk >> 2;
                const int koff  = (chunk & 3) * 8;
                *(h8*)(&sB[cur ^ 1][row * 32 + koff]) = pre[c];
            }
        }
        __syncthreads();
    }
#endif

    // C/D layout: vgpr r -> M = r + 8*(lane>=16), N = lane%16
    const int mo = (lane >> 4) * 8;
    #pragma unroll
    for (int j = 0; j < NT; ++j) {
        const int n = n0 + 16 * j + mrow;
        #pragma unroll
        for (int r = 0; r < 8; ++r) {
            float v = acc[j][r];
            if (EPI == 1) v = softplus_eps(v + bias[n]);
            C[(size_t)(m0 + mo + r) * ldc + n] = v;
        }
    }
}

// ---------------------------------------------------------------------------
// Depthwise causal conv1d (width 4) + SiLU.  Writes f32 (for scan) + f16
// (for the x_proj WMMA GEMM).
// ---------------------------------------------------------------------------
__global__ __launch_bounds__(256)
void conv_silu(const float* __restrict__ xz, const float* __restrict__ Wc,
               const float* __restrict__ bc, float* __restrict__ xc,
               _Float16* __restrict__ xc16)
{
    const int idx = blockIdx.x * blockDim.x + threadIdx.x;
    if (idx >= NTOK * D_INNER) return;
    const int c = idx % D_INNER;
    const int t = idx / D_INNER;
    const int s = t % SEQ;
    float acc = bc[c];
    #pragma unroll
    for (int k = 0; k < D_CONV; ++k) {
        const int sp = s - (D_CONV - 1) + k;
        if (sp >= 0)
            acc += Wc[c * D_CONV + k] * xz[(size_t)(t - (D_CONV - 1 - k)) * (2 * D_INNER) + c];
    }
    const float v = acc / (1.0f + __expf(-acc));   // SiLU
    xc[idx]   = v;
    xc16[idx] = (_Float16)v;
}

// ---------------------------------------------------------------------------
// Selective scan: h_t = dA_t h_{t-1} + dt B_t u_t ; y_t = <h_t, C_t>.
// One thread per (batch, channel); 16 states in registers; per-token B/C
// staged through LDS.  Fused +u*D skip and SiLU(z) gate.
// ---------------------------------------------------------------------------
__global__ __launch_bounds__(64)
void selective_scan(const float* __restrict__ xz, const float* __restrict__ xc,
                    const float* __restrict__ dbl, const float* __restrict__ dtb,
                    const float* __restrict__ A_log, const float* __restrict__ Dv,
                    float* __restrict__ yg)
{
    const int b = blockIdx.x / (D_INNER / 64);
    const int d = (blockIdx.x % (D_INNER / 64)) * 64 + threadIdx.x;

    __shared__ float sBC[2 * D_STATE];

    float Arow[D_STATE];
    #pragma unroll
    for (int n = 0; n < D_STATE; ++n) Arow[n] = -__expf(A_log[(size_t)d * D_STATE + n]);
    const float Dd = Dv[d];

    float h[D_STATE];
    #pragma unroll
    for (int n = 0; n < D_STATE; ++n) h[n] = 0.0f;

    for (int s = 0; s < SEQ; ++s) {
        const int t = b * SEQ + s;
        if (threadIdx.x < 2 * D_STATE)
            sBC[threadIdx.x] = dbl[(size_t)t * XPROJ_N + DT_RANK + threadIdx.x];
        __syncthreads();

        const float dtv = dtb[(size_t)t * D_INNER + d];
        const float u   = xc[(size_t)t * D_INNER + d];
        float y = 0.0f;
        #pragma unroll
        for (int n = 0; n < D_STATE; ++n) {
            float e = dtv * Arow[n];
            e = fminf(10.0f, fmaxf(-10.0f, e));
            const float dA = __expf(e);
            h[n] = dA * h[n] + (dtv * sBC[n]) * u;
            y   += h[n] * sBC[D_STATE + n];
        }
        const float z = xz[(size_t)t * (2 * D_INNER) + D_INNER + d];
        const float g = z / (1.0f + __expf(-z));
        yg[(size_t)t * D_INNER + d] = (y + u * Dd) * g;
        __syncthreads();
    }
}

// ---------------------------------------------------------------------------
// RMSNorm over D_INNER per token (eps 1e-6), scaled by rms_w; emits f16
// directly for the out_proj WMMA GEMM.
// ---------------------------------------------------------------------------
__global__ __launch_bounds__(256)
void rmsnorm(const float* __restrict__ yg, const float* __restrict__ w,
             _Float16* __restrict__ yn16)
{
    const int t = blockIdx.x;
    const float* __restrict__ row = yg + (size_t)t * D_INNER;
    __shared__ float red[8];

    float ss = 0.0f;
    for (int c = threadIdx.x; c < D_INNER; c += 256) { const float v = row[c]; ss += v * v; }
    #pragma unroll
    for (int off = 16; off > 0; off >>= 1) ss += __shfl_down(ss, off, 32);
    if ((threadIdx.x & 31) == 0) red[threadIdx.x >> 5] = ss;
    __syncthreads();
    if (threadIdx.x == 0) {
        float tot = 0.0f;
        #pragma unroll
        for (int i = 0; i < 8; ++i) tot += red[i];
        red[0] = rsqrtf(tot / (float)D_INNER + 1e-6f);
    }
    __syncthreads();
    const float scale = red[0];
    for (int c = threadIdx.x; c < D_INNER; c += 256)
        yn16[(size_t)t * D_INNER + c] = (_Float16)(row[c] * scale * w[c]);
}

// ---------------------------------------------------------------------------
// Host-side launch
// ---------------------------------------------------------------------------
extern "C" void kernel_launch(void* const* d_in, const int* in_sizes, int n_in,
                              void* d_out, int out_size, void* d_ws, size_t ws_size,
                              hipStream_t stream) {
    const float* x      = (const float*)d_in[0];   // [2,1024,1024]
    const float* W_in   = (const float*)d_in[1];   // [4096,1024]
    const float* W_conv = (const float*)d_in[2];   // [2048,4]
    const float* b_conv = (const float*)d_in[3];   // [2048]
    const float* W_x    = (const float*)d_in[4];   // [96,2048]
    const float* W_dt   = (const float*)d_in[5];   // [2048,64]
    const float* b_dt   = (const float*)d_in[6];   // [2048]
    const float* A_log  = (const float*)d_in[7];   // [2048,16]
    const float* Dv     = (const float*)d_in[8];   // [2048]
    const float* rms_w  = (const float*)d_in[9];   // [2048]
    const float* W_out  = (const float*)d_in[10];  // [1024,2048]
    float* out = (float*)d_out;                    // [2,1024,1024]

    // workspace carve-out (256B aligned chunks)
    char* p = (char*)d_ws;
    auto carve = [&](size_t bytes) { void* r = (void*)p; p += (bytes + 255) & ~(size_t)255; return r; };

    float*     xz     = (float*)    carve((size_t)NTOK * 2 * D_INNER * 4);
    float*     xc     = (float*)    carve((size_t)NTOK * D_INNER * 4);
    float*     dbl    = (float*)    carve((size_t)NTOK * XPROJ_N * 4);
    float*     dtb    = (float*)    carve((size_t)NTOK * D_INNER * 4);
    float*     yg     = (float*)    carve((size_t)NTOK * D_INNER * 4);
    _Float16*  x16    = (_Float16*) carve((size_t)NTOK * DIM * 2);
    _Float16*  Win16  = (_Float16*) carve((size_t)2 * D_INNER * DIM * 2);
    _Float16*  Wx16   = (_Float16*) carve((size_t)XPROJ_N * D_INNER * 2);
    _Float16*  Wdt16  = (_Float16*) carve((size_t)D_INNER * DT_RANK * 2);
    _Float16*  Wout16 = (_Float16*) carve((size_t)DIM * D_INNER * 2);
    _Float16*  xc16   = (_Float16*) carve((size_t)NTOK * D_INNER * 2);
    _Float16*  dbl16  = (_Float16*) carve((size_t)NTOK * XPROJ_N * 2);
    _Float16*  yn16   = (_Float16*) carve((size_t)NTOK * D_INNER * 2);

    auto cvt = [&](const float* s, _Float16* d, size_t n) {
        cvt_f32_f16<<<(unsigned)((n / 8 + 255) / 256), 256, 0, stream>>>(s, d, n);
    };

    // 0) one-shot f16 conversions of input activations + weights
    cvt(x,     x16,    (size_t)NTOK * DIM);
    cvt(W_in,  Win16,  (size_t)2 * D_INNER * DIM);
    cvt(W_x,   Wx16,   (size_t)XPROJ_N * D_INNER);
    cvt(W_dt,  Wdt16,  (size_t)D_INNER * DT_RANK);
    cvt(W_out, Wout16, (size_t)DIM * D_INNER);

    // 1) in_proj: xz = x @ W_in^T            M=2048 N=4096 K=1024  (NT=4)
    gemm_wmma<4, 0><<<dim3(NTOK / 64, (2 * D_INNER) / 64), 128, 0, stream>>>(
        x16, Win16, xz, nullptr, DIM, DIM, DIM, 2 * D_INNER);

    // 2) depthwise conv + SiLU (f32 + f16 outputs)
    conv_silu<<<(NTOK * D_INNER + 255) / 256, 256, 0, stream>>>(xz, W_conv, b_conv, xc, xc16);

    // 3) x_proj: dbl = xc @ W_x^T            M=2048 N=96 K=2048   (NT=2, grid.y=3)
    gemm_wmma<2, 0><<<dim3(NTOK / 64, XPROJ_N / 32), 128, 0, stream>>>(
        xc16, Wx16, dbl, nullptr, D_INNER, D_INNER, D_INNER, XPROJ_N);

    // 3b) f16 copy of dbl for the dt GEMM
    cvt(dbl, dbl16, (size_t)NTOK * XPROJ_N);

    // 4) dt_proj + softplus: dt = softplus(dbl[:, :64] @ W_dt^T + b_dt) + 1e-4  (NT=4)
    gemm_wmma<4, 1><<<dim3(NTOK / 64, D_INNER / 64), 128, 0, stream>>>(
        dbl16, Wdt16, dtb, b_dt, DT_RANK, XPROJ_N, DT_RANK, D_INNER);

    // 5) selective scan + D-skip + SiLU(z) gate
    selective_scan<<<BATCH * (D_INNER / 64), 64, 0, stream>>>(
        xz, xc, dbl, dtb, A_log, Dv, yg);

    // 6) RMSNorm (emits f16)
    rmsnorm<<<NTOK, 256, 0, stream>>>(yg, rms_w, yn16);

    // 7) out_proj: out = yn @ W_out^T        M=2048 N=1024 K=2048  (NT=4)
    gemm_wmma<4, 0><<<dim3(NTOK / 64, DIM / 64), 128, 0, stream>>>(
        yn16, Wout16, out, nullptr, D_INNER, D_INNER, D_INNER, DIM);
}
